// PCT_5746666242093
// MI455X (gfx1250) — matrix operand order repaired and checked by reading, not compiled, not run on hardware
//
#include <hip/hip_runtime.h>

// ---------------- CDNA5 (gfx1250) WMMA types ----------------
typedef __attribute__((ext_vector_type(16))) __bf16 v16bf;
typedef __attribute__((ext_vector_type(8)))  float  v8f;

#define B_   16
#define C_   256
#define N_   2048
#define D_   64
#define CAT_ 1280
#define OUT_ 1024

typedef unsigned short bf16u;   // raw bf16 storage

__device__ __forceinline__ bf16u f2bf(float f) {
  unsigned int u = __float_as_uint(f);
  u += 0x7fffu + ((u >> 16) & 1u);          // round-to-nearest-even
  return (bf16u)(u >> 16);
}

// Two contiguous 16B loads -> one 16-element bf16 fragment
__device__ __forceinline__ v16bf load_frag(const bf16u* p0, const bf16u* p1) {
  union { v16bf v; float4 f[2]; } u;
  u.f[0] = *reinterpret_cast<const float4*>(p0);
  u.f[1] = *reinterpret_cast<const float4*>(p1);
  return u.v;
}

__device__ __forceinline__ v8f wmma_bf16(v16bf a, v16bf b, v8f c) {
  return __builtin_amdgcn_wmma_f32_16x16x32_bf16(false, a, false, b, (short)0, c,
                                                 false, false);
}

// =====================================================================
// Generic bf16 WMMA GEMM:  out[M,N] = A[M,K] * Bt[N,K]^T   (per batch z)
// A row-major bf16 (lda=K), Bt row-major bf16 (ldb=K).
// Wave computes a 16x64 strip (4 accumulators, A reuse). Block = 8 waves
// = 32x256 tile. All M multiples of 32, N==2048, K multiple of 64.
// Inner loop is software-pipelined: fragments for k-step i+1 are loaded
// into a second register buffer before the WMMAs of step i, so global
// loads overlap the XDL pipe instead of draining loadcnt to 0 each step.
// EPI: 0 = f32 out (+opt per-row bias)
//      1 = bf16 out row-major (+opt per-row bias)
//      2 = bf16 out TRANSPOSED (out[n*M + m])
//      3 = f32 out, val*scale[row]+shift[row], LeakyReLU(slope)
// =====================================================================
template<int EPI>
__global__ void __launch_bounds__(256) gemm_bf16_wmma(
    const bf16u* __restrict__ A,  long long sA,
    const bf16u* __restrict__ Bt, long long sBt,
    void* __restrict__ outp,      long long sOut,
    int M, int N, int K,
    const float* __restrict__ bias,
    const float* __restrict__ scale,
    const float* __restrict__ shift,
    float slope)
{
  const int lane = threadIdx.x & 31;
  const int wave = threadIdx.x >> 5;
  const int z    = blockIdx.z;
  const int hl   = lane & 15;
  const int hi   = lane >> 4;

  const int tileM = blockIdx.y * 32  + (wave >> 2) * 16;
  const int tileN = blockIdx.x * 256 + (wave & 3) * 64;

  const bf16u* Ab = A  + (size_t)z * (size_t)sA;
  const bf16u* Bb = Bt + (size_t)z * (size_t)sBt;
  const bf16u* arow = Ab + (size_t)(tileM + hl) * (size_t)K;

  // A fragment: lane hl holds row (tileM+hl); K = (e&7) + 8*(2*(e>=8)+hi)
  auto ldA = [&](int kb) -> v16bf {
    return load_frag(arow + kb + 8 * hi, arow + kb + 16 + 8 * hi);
  };
  // B fragment: column n = tileN+16j+hl ; K = (e&7) + 8*(2*(e>=8)+hi)
  auto ldB = [&](int j, int kb) -> v16bf {
    const bf16u* p = Bb + (size_t)(tileN + 16 * j + hl) * (size_t)K + kb + 8 * hi;
    return load_frag(p, p + 16);
  };

  v8f acc[4] = {};

  // ---- software-pipelined main loop (ping-pong register buffers) ----
  v16bf a0 = ldA(0);
  v16bf b00 = ldB(0, 0), b01 = ldB(1, 0), b02 = ldB(2, 0), b03 = ldB(3, 0);

  for (int kb = 0; kb < K; kb += 64) {
    const int k1 = kb + 32;                       // K % 64 == 0  ->  k1 < K
    v16bf a1  = ldA(k1);
    v16bf b10 = ldB(0, k1), b11 = ldB(1, k1), b12 = ldB(2, k1), b13 = ldB(3, k1);

    acc[0] = wmma_bf16(a0, b00, acc[0]);
    acc[1] = wmma_bf16(a0, b01, acc[1]);
    acc[2] = wmma_bf16(a0, b02, acc[2]);
    acc[3] = wmma_bf16(a0, b03, acc[3]);

    const int k2 = (kb + 64 < K) ? kb + 64 : 0;   // clamped dummy prefetch at end
    a0  = ldA(k2);
    b00 = ldB(0, k2); b01 = ldB(1, k2); b02 = ldB(2, k2); b03 = ldB(3, k2);

    acc[0] = wmma_bf16(a1, b10, acc[0]);
    acc[1] = wmma_bf16(a1, b11, acc[1]);
    acc[2] = wmma_bf16(a1, b12, acc[2]);
    acc[3] = wmma_bf16(a1, b13, acc[3]);
  }

  // ---- epilogue: C/D layout is row = tileM + r + 8*hi, col = tileN+16j+hl ----
  const int row0 = tileM + hi * 8;
#pragma unroll
  for (int j = 0; j < 4; ++j) {
    const int col = tileN + 16 * j + hl;
#pragma unroll
    for (int r = 0; r < 8; ++r) {
      const int row = row0 + r;
      float v = acc[j][r];
      if (EPI == 0) {
        if (bias) v += bias[row];
        ((float*)outp)[(size_t)z * sOut + (size_t)row * N + col] = v;
      } else if (EPI == 1) {
        if (bias) v += bias[row];
        ((bf16u*)outp)[(size_t)z * sOut + (size_t)row * N + col] = f2bf(v);
      } else if (EPI == 2) {
        ((bf16u*)outp)[(size_t)z * sOut + (size_t)col * M + row] = f2bf(v);
      } else {
        v = v * scale[row] + shift[row];
        v = (v >= 0.f) ? v : slope * v;
        ((float*)outp)[(size_t)z * sOut + (size_t)row * N + col] = v;
      }
    }
  }
}

// ---------------- block reductions ----------------
__device__ __forceinline__ float blk_max(float v, float* sm) {
  int t = threadIdx.x;
  sm[t] = v; __syncthreads();
  for (int s = 128; s > 0; s >>= 1) { if (t < s) sm[t] = fmaxf(sm[t], sm[t + s]); __syncthreads(); }
  float r = sm[0]; __syncthreads();
  return r;
}
__device__ __forceinline__ float blk_sum(float v, float* sm) {
  int t = threadIdx.x;
  sm[t] = v; __syncthreads();
  for (int s = 128; s > 0; s >>= 1) { if (t < s) sm[t] += sm[t + s]; __syncthreads(); }
  float r = sm[0]; __syncthreads();
  return r;
}

// ---------------- elementwise / softmax / pooling kernels ----------------
__global__ void cvt_f32_bf16(const float* __restrict__ in, bf16u* __restrict__ out, size_t n) {
  size_t i = (size_t)blockIdx.x * 256 + threadIdx.x;
  if (i < n) out[i] = f2bf(in[i]);
}

// gamma/beta(/conv-bias) -> effective per-channel scale & shift for fused BN
__global__ void prep_eff(const float* __restrict__ g, const float* __restrict__ be,
                         const float* __restrict__ cb, float* __restrict__ s,
                         float* __restrict__ sh, int n) {
  int i = blockIdx.x * 256 + threadIdx.x;
  if (i >= n) return;
  float sc = g[i] * rsqrtf(1.f + 1e-5f);
  s[i]  = sc;
  sh[i] = be[i] + (cb ? cb[i] * sc : 0.f);
}

// h = x ; hT/catT slot0 = bf16(x) transposed
__global__ void init_feats(const float* __restrict__ x, float* __restrict__ h,
                           bf16u* __restrict__ hT, bf16u* __restrict__ catT) {
  size_t idx = (size_t)blockIdx.x * 256 + threadIdx.x;
  if (idx >= (size_t)B_ * C_ * N_) return;
  int n = (int)(idx % N_);
  size_t t = idx / N_;
  int c = (int)(t % C_);
  int b = (int)(t / C_);
  float v = x[idx];
  h[idx] = v;
  bf16u bv = f2bf(v);
  hT[((size_t)b * N_ + n) * C_ + c]     = bv;
  catT[((size_t)b * N_ + n) * CAT_ + c] = bv;
}

// rowmax / rowsum(exp) of each 2048-wide row of E
__global__ void row_stats(const float* __restrict__ E, float* __restrict__ rmax,
                          float* __restrict__ rsum) {
  __shared__ float sm[256];
  size_t rowi = (size_t)blockIdx.y * N_ + blockIdx.x;
  const float* row = E + rowi * N_;
  float m = -3.4e38f;
  for (int i = threadIdx.x; i < N_; i += 256) m = fmaxf(m, row[i]);
  m = blk_max(m, sm);
  float s = 0.f;
  for (int i = threadIdx.x; i < N_; i += 256) s += __expf(row[i] - m);
  s = blk_sum(s, sm);
  if (threadIdx.x == 0) { rmax[rowi] = m; rsum[rowi] = s; }
}

// attT[m][n] = exp(E[m][n]-rmax[n])/rsum[n]  (E symmetric => this is att[n][m]),
// then divide whole row by (1e-9 + row-sum)  == the reference's column-sum norm.
__global__ void softmax_dn(const float* __restrict__ E, const float* __restrict__ rmax,
                           const float* __restrict__ rsum, bf16u* __restrict__ attT) {
  __shared__ float sm[256];
  const int z = blockIdx.y, m = blockIdx.x;
  const float* Erow = E + ((size_t)z * N_ + m) * N_;
  const float* rm = rmax + (size_t)z * N_;
  const float* rs = rsum + (size_t)z * N_;
  float v[8]; float loc = 0.f;
#pragma unroll
  for (int j = 0; j < 8; ++j) {
    int n = threadIdx.x + j * 256;
    v[j] = __expf(Erow[n] - rm[n]) / rs[n];
    loc += v[j];
  }
  float S = blk_sum(loc, sm);
  float sc = 1.f / (1e-9f + S);
  bf16u* arow = attT + ((size_t)z * N_ + m) * N_;
#pragma unroll
  for (int j = 0; j < 8; ++j) arow[threadIdx.x + j * 256] = f2bf(v[j] * sc);
}

// dT[b][n][c] = bf16(h - xr)
__global__ void make_dT(const float* __restrict__ h, const float* __restrict__ xr,
                        bf16u* __restrict__ dT) {
  size_t idx = (size_t)blockIdx.x * 256 + threadIdx.x;
  if (idx >= (size_t)B_ * C_ * N_) return;
  int n = (int)(idx % N_);
  size_t t = idx / N_;
  int c = (int)(t % C_);
  int b = (int)(t / C_);
  dT[((size_t)b * N_ + n) * C_ + c] = f2bf(h[idx] - xr[idx]);
}

// h += t (t already relu(bn(..))); refresh hT and catT slot
__global__ void update_h(const float* __restrict__ tF, float* __restrict__ h,
                         bf16u* __restrict__ hT, bf16u* __restrict__ catT, int slot) {
  size_t idx = (size_t)blockIdx.x * 256 + threadIdx.x;
  if (idx >= (size_t)B_ * C_ * N_) return;
  int n = (int)(idx % N_);
  size_t t = idx / N_;
  int c = (int)(t % C_);
  int b = (int)(t / C_);
  float hn = h[idx] + tF[idx];
  h[idx] = hn;
  bf16u bv = f2bf(hn);
  hT[((size_t)b * N_ + n) * C_ + c] = bv;
  catT[((size_t)b * N_ + n) * CAT_ + slot * C_ + c] = bv;
}

// max & mean over last axis of y[B,OUT,N]
__global__ void pool_maxmean(const float* __restrict__ y, float* __restrict__ xmax,
                             float* __restrict__ xmean) {
  __shared__ float sm[256];
  size_t rowi = (size_t)blockIdx.y * OUT_ + blockIdx.x;
  const float* row = y + rowi * N_;
  float m = -3.4e38f, s = 0.f;
  for (int i = threadIdx.x; i < N_; i += 256) { float v = row[i]; m = fmaxf(m, v); s += v; }
  m = blk_max(m, sm);
  s = blk_sum(s, sm);
  if (threadIdx.x == 0) { xmax[rowi] = m; xmean[rowi] = s * (1.f / N_); }
}

// =====================================================================
extern "C" void kernel_launch(void* const* d_in, const int* in_sizes, int n_in,
                              void* d_out, int out_size, void* d_ws, size_t ws_size,
                              hipStream_t stream) {
  (void)in_sizes; (void)n_in; (void)out_size; (void)ws_size;
  const float* x     = (const float*)d_in[0];
  const float* Wqk   = (const float*)d_in[1];
  const float* Wv    = (const float*)d_in[2];
  const float* bv    = (const float*)d_in[3];
  const float* Wt    = (const float*)d_in[4];
  const float* bt    = (const float*)d_in[5];
  const float* gamma = (const float*)d_in[6];
  const float* beta  = (const float*)d_in[7];
  const float* Wlin  = (const float*)d_in[8];
  const float* glin  = (const float*)d_in[9];
  const float* blin  = (const float*)d_in[10];
  float* out = (float*)d_out;

  char* ws = (char*)d_ws;
  size_t off = 0;
  auto alloc = [&](size_t bytes) -> void* {
    off = (off + 255) & ~(size_t)255;
    void* p = ws + off; off += bytes; return p;
  };
  bf16u* WqkB  = (bf16u*)alloc((size_t)4 * D_ * C_ * 2);
  bf16u* WvB   = (bf16u*)alloc((size_t)4 * C_ * C_ * 2);
  bf16u* WtB   = (bf16u*)alloc((size_t)4 * C_ * C_ * 2);
  bf16u* WlinB = (bf16u*)alloc((size_t)OUT_ * CAT_ * 2);
  bf16u* hT    = (bf16u*)alloc((size_t)B_ * N_ * C_ * 2);
  float* hF    = (float*)alloc((size_t)B_ * C_ * N_ * 4);
  bf16u* qT    = (bf16u*)alloc((size_t)B_ * N_ * D_ * 2);
  bf16u* vB    = (bf16u*)alloc((size_t)B_ * C_ * N_ * 2);
  float* E     = (float*)alloc((size_t)B_ * N_ * N_ * 4);
  float* rmax  = (float*)alloc((size_t)B_ * N_ * 4);
  float* rsum  = (float*)alloc((size_t)B_ * N_ * 4);
  bf16u* attT  = (bf16u*)alloc((size_t)B_ * N_ * N_ * 2);
  float* xr    = (float*)alloc((size_t)B_ * C_ * N_ * 4);
  bf16u* dT    = (bf16u*)alloc((size_t)B_ * N_ * C_ * 2);
  float* tF    = (float*)alloc((size_t)B_ * C_ * N_ * 4);
  float* effS  = (float*)alloc((size_t)OUT_ * 4);
  float* effB  = (float*)alloc((size_t)OUT_ * 4);
  bf16u* catT  = (bf16u*)alloc((size_t)B_ * N_ * CAT_ * 2);

  auto cdiv = [](size_t a, size_t b) { return (unsigned)((a + b - 1) / b); };
  const size_t BCN = (size_t)B_ * C_ * N_;

  // weights -> bf16
  cvt_f32_bf16<<<cdiv((size_t)4 * D_ * C_, 256), 256, 0, stream>>>(Wqk, WqkB, (size_t)4 * D_ * C_);
  cvt_f32_bf16<<<cdiv((size_t)4 * C_ * C_, 256), 256, 0, stream>>>(Wv, WvB, (size_t)4 * C_ * C_);
  cvt_f32_bf16<<<cdiv((size_t)4 * C_ * C_, 256), 256, 0, stream>>>(Wt, WtB, (size_t)4 * C_ * C_);
  cvt_f32_bf16<<<cdiv((size_t)OUT_ * CAT_, 256), 256, 0, stream>>>(Wlin, WlinB, (size_t)OUT_ * CAT_);

  init_feats<<<cdiv(BCN, 256), 256, 0, stream>>>(x, hF, hT, catT);

  for (int i = 0; i < 4; ++i) {
    prep_eff<<<1, 256, 0, stream>>>(gamma + i * C_, beta + i * C_, bt + i * C_, effS, effB, C_);
    // q^T [N,D] = (Wqk[i] [D,C] x h [C,N]) transposed-store
    gemm_bf16_wmma<2><<<dim3(N_ / 256, D_ / 32, B_), 256, 0, stream>>>(
        WqkB + (size_t)i * D_ * C_, 0LL, hT, (long long)N_ * C_,
        qT, (long long)N_ * D_, D_, N_, C_, nullptr, nullptr, nullptr, 0.f);
    // v [C,N] bf16 = Wv[i] x h + bv
    gemm_bf16_wmma<1><<<dim3(N_ / 256, C_ / 32, B_), 256, 0, stream>>>(
        WvB + (size_t)i * C_ * C_, 0LL, hT, (long long)N_ * C_,
        vB, (long long)C_ * N_, C_, N_, C_, bv + i * C_, nullptr, nullptr, 0.f);
    // energy [N,N] = qT x qT^T   (symmetric)
    gemm_bf16_wmma<0><<<dim3(N_ / 256, N_ / 32, B_), 256, 0, stream>>>(
        qT, (long long)N_ * D_, qT, (long long)N_ * D_,
        E, (long long)N_ * N_, N_, N_, D_, nullptr, nullptr, nullptr, 0.f);
    row_stats<<<dim3(N_, B_), 256, 0, stream>>>(E, rmax, rsum);
    softmax_dn<<<dim3(N_, B_), 256, 0, stream>>>(E, rmax, rsum, attT);
    // x_r [C,N] = v x att   (attT is the transposed-B operand)
    gemm_bf16_wmma<0><<<dim3(N_ / 256, C_ / 32, B_), 256, 0, stream>>>(
        vB, (long long)C_ * N_, attT, (long long)N_ * N_,
        xr, (long long)C_ * N_, C_, N_, N_, nullptr, nullptr, nullptr, 0.f);
    make_dT<<<cdiv(BCN, 256), 256, 0, stream>>>(hF, xr, dT);
    // t = relu(bn(Wt x (h - x_r) + bt))  -- BN+bias+ReLU fused, slope 0
    gemm_bf16_wmma<3><<<dim3(N_ / 256, C_ / 32, B_), 256, 0, stream>>>(
        WtB + (size_t)i * C_ * C_, 0LL, dT, (long long)N_ * C_,
        tF, (long long)C_ * N_, C_, N_, C_, nullptr, effS, effB, 0.f);
    update_h<<<cdiv(BCN, 256), 256, 0, stream>>>(tF, hF, hT, catT, i + 1);
  }

  // y = LeakyReLU(bn(Wlin x cat)) straight into d_out
  prep_eff<<<4, 256, 0, stream>>>(glin, blin, nullptr, effS, effB, OUT_);
  gemm_bf16_wmma<3><<<dim3(N_ / 256, OUT_ / 32, B_), 256, 0, stream>>>(
      WlinB, 0LL, catT, (long long)N_ * CAT_,
      out, (long long)OUT_ * N_, OUT_, N_, CAT_, nullptr, effS, effB, 0.2f);

  float* xmax  = out + (size_t)B_ * OUT_ * N_;
  float* xmean = xmax + (size_t)B_ * OUT_;
  pool_maxmean<<<dim3(OUT_, B_), 256, 0, stream>>>(out, xmax, xmean);
}
